// GatedLinearRecurrence_34316788695877
// MI455X (gfx1250) — matrix operand order repaired and checked by reading, not compiled, not run on hardware
//
#include <hip/hip_runtime.h>
#include <hip/hip_bf16.h>

// ---------------------------------------------------------------------------
// Problem constants (from reference)
// ---------------------------------------------------------------------------
constexpr int Bb    = 2;
constexpr int T     = 4096;
constexpr int Dd    = 2048;
constexpr int H     = 16;
constexpr int DH    = 128;
constexpr int HID   = H * DH;       // 2048
constexpr int CHUNK = 64;
constexpr int NC    = T / CHUNK;    // 64
constexpr int MROWS = Bb * T;       // 8192

typedef __attribute__((ext_vector_type(16))) __bf16       v16bf;
typedef __attribute__((ext_vector_type(8)))  float        v8f;
typedef __attribute__((ext_vector_type(4)))  unsigned int u32x4;

// 32-bit LDS offset of a generic pointer known to be LDS (low 32 bits of the
// flat address are the LDS byte address on CDNA5).
typedef __attribute__((address_space(3))) char lds_char;
__device__ __forceinline__ unsigned lds_off(void* p) {
  return (unsigned)(unsigned long long)(lds_char*)p;
}

// ---------------------------------------------------------------------------
// fp32 -> bf16 elementwise convert
// ---------------------------------------------------------------------------
__global__ void f32_to_bf16_kernel(const float* __restrict__ in,
                                   __bf16* __restrict__ out, size_t n) {
  size_t i = (size_t)blockIdx.x * blockDim.x + threadIdx.x;
  if (i < n) out[i] = (__bf16)in[i];
}

// ---------------------------------------------------------------------------
// fp32 (K x N) -> bf16 transposed (N x K), 32x32 tiles through LDS
// ---------------------------------------------------------------------------
__global__ __launch_bounds__(256)
void transpose_f32_to_bf16_kernel(const float* __restrict__ W,
                                  __bf16* __restrict__ Wt, int K, int N) {
  __shared__ float tile[32][33];
  const int nb = blockIdx.x * 32;
  const int kb = blockIdx.y * 32;
  const int tx = threadIdx.x & 31;
  const int tg = threadIdx.x >> 5;   // 0..7
#pragma unroll
  for (int i = 0; i < 32; i += 8)
    tile[tg + i][tx] = W[(size_t)(kb + tg + i) * N + nb + tx];
  __syncthreads();
#pragma unroll
  for (int i = 0; i < 32; i += 8)
    Wt[(size_t)(nb + tg + i) * K + kb + tx] = (__bf16)tile[tx][tg + i];
}

// ---------------------------------------------------------------------------
// WMMA bf16 GEMM:  C[M,N] = A[M,K] * Bt[N,K]^T   (fp32 accumulate/output)
// Block: 128 threads = 4 waves. Block tile 128x128, wave tile 64x64,
// K-step 32. Double-buffered LDS fed by GLOBAL_LOAD_ASYNC_TO_LDS_B128.
// K is a compile-time constant so the ASYNCcnt pipeline has no branches in
// the steady-state loop; the drain (s_wait_asynccnt 0) is peeled.
// ---------------------------------------------------------------------------
constexpr int BM  = 128;
constexpr int BN  = 128;
constexpr int BK  = 32;
constexpr int LDT = BK + 16;   // LDS row stride in halves (96B, 16B aligned)

__device__ __forceinline__ v16bf load_frag_b128x2(const __bf16* p, int second_off_halves) {
  union { u32x4 q[2]; v16bf v; } t;
  t.q[0] = *reinterpret_cast<const u32x4*>(p);
  t.q[1] = *reinterpret_cast<const u32x4*>(p + second_off_halves);
  return t.v;
}

// One 64-byte row-chunk: 4 async b128 copies. IOFFSET advances BOTH the LDS
// destination and the global source (ISA 10.x async pseudocode), so a single
// (lds, gaddr) pair covers the contiguous 64B.
__device__ __forceinline__ void async_copy_row64(unsigned lds, unsigned long long ga) {
  asm volatile("global_load_async_to_lds_b128 %0, %1, off"
               :: "v"(lds), "v"(ga) : "memory");
  asm volatile("global_load_async_to_lds_b128 %0, %1, off offset:16"
               :: "v"(lds), "v"(ga) : "memory");
  asm volatile("global_load_async_to_lds_b128 %0, %1, off offset:32"
               :: "v"(lds), "v"(ga) : "memory");
  asm volatile("global_load_async_to_lds_b128 %0, %1, off offset:48"
               :: "v"(lds), "v"(ga) : "memory");
}

template <int K>
__global__ __launch_bounds__(128)
void wmma_gemm_bf16_kernel(const __bf16* __restrict__ A,
                           const __bf16* __restrict__ Bt,
                           float* __restrict__ C,
                           int M, int N) {
  static_assert(K % (2 * BK) == 0, "K must be a multiple of 64");
  __shared__ __align__(16) __bf16 As[2][BM * LDT];
  __shared__ __align__(16) __bf16 Bs[2][BN * LDT];

  const int tid   = threadIdx.x;
  const int wave  = tid >> 5;
  const int lane  = tid & 31;
  const int laneN = lane & 15;       // row (A frag) / column (B frag) in tile
  const int laneH = lane >> 4;       // K-half selector
  const int wm    = (wave >> 1) * 64;
  const int wn    = (wave & 1)  * 64;
  const int m0    = blockIdx.y * BM;
  const int n0    = blockIdx.x * BN;

  const v8f vzero = {0.f, 0.f, 0.f, 0.f, 0.f, 0.f, 0.f, 0.f};
  v8f acc[4][4];
#pragma unroll
  for (int i = 0; i < 4; ++i)
#pragma unroll
    for (int j = 0; j < 4; ++j) acc[i][j] = vzero;

  const __bf16* aRow = A  + (size_t)(m0 + tid) * K;   // one row per thread
  const __bf16* bRow = Bt + (size_t)(n0 + tid) * K;
  const unsigned ldsA[2] = { lds_off(&As[0][tid * LDT]), lds_off(&As[1][tid * LDT]) };
  const unsigned ldsB[2] = { lds_off(&Bs[0][tid * LDT]), lds_off(&Bs[1][tid * LDT]) };

  auto issue_tile = [&](int kb, int s) {
    async_copy_row64(ldsA[s], (unsigned long long)(aRow + kb));
    async_copy_row64(ldsB[s], (unsigned long long)(bRow + kb));
  };

  auto compute_stage = [&](int s) {
    // Fragments (ISA 7.12.2):
    //  A (16x32 bf16): lane = row M, K chunks at halves 8*laneH and 16+8*laneH
    //  B (32x16 bf16): lane = col N, K chunks at halves 8*laneH and 16+8*laneH
    v16bf af[4], bfm[4];
#pragma unroll
    for (int i = 0; i < 4; ++i)
      af[i]  = load_frag_b128x2(&As[s][(wm + i * 16 + laneN) * LDT + 8 * laneH], 16);
#pragma unroll
    for (int j = 0; j < 4; ++j)
      bfm[j] = load_frag_b128x2(&Bs[s][(wn + j * 16 + laneN) * LDT + 8 * laneH], 16);

#pragma unroll
    for (int i = 0; i < 4; ++i)
#pragma unroll
      for (int j = 0; j < 4; ++j)
        acc[i][j] = __builtin_amdgcn_wmma_f32_16x16x32_bf16(
            /*neg_a=*/false, af[i], /*neg_b=*/false, bfm[j],
            /*c_mod=*/(short)0, acc[i][j],
            /*reuse_a=*/false, /*reuse_b=*/false);
  };

  // One pipeline step: optionally prefetch the next tile, wait until the
  // current tile's 8 async ops (the oldest outstanding, in-order retirement)
  // have landed, then compute.
  auto step_next = [&](int kb, int sComp, int sNext) {
    issue_tile(kb + BK, sNext);
    asm volatile("s_wait_asynccnt 0x8" ::: "memory");
    __syncthreads();
    compute_stage(sComp);
    __syncthreads();
  };
  auto step_last = [&](int sComp) {
    asm volatile("s_wait_asynccnt 0x0" ::: "memory");
    __syncthreads();
    compute_stage(sComp);
    // no trailing barrier needed: nothing overwrites LDS after this
  };

  issue_tile(0, 0);           // prologue: tile 0 -> stage 0
  int kb = 0;
#pragma unroll 1
  for (; kb < K - 2 * BK; kb += 2 * BK) {   // branch-free steady state
    step_next(kb,      0, 1);
    step_next(kb + BK, 1, 0);
  }
  step_next(kb, 0, 1);        // prefetches the final tile (K-BK)
  step_last(1);               // drain

  // ---- epilogue: C/D layout -> lane col = laneN, row = vgpr + 8*laneH ----
#pragma unroll
  for (int i = 0; i < 4; ++i)
#pragma unroll
    for (int j = 0; j < 4; ++j) {
      float* cp = C + (size_t)(m0 + wm + i * 16 + 8 * laneH) * N
                    + (n0 + wn + j * 16 + laneN);
#pragma unroll
      for (int r = 0; r < 8; ++r) cp[(size_t)r * N] = acc[i][j][r];
    }
}

// ---------------------------------------------------------------------------
// Decay: a[row,h] = sigmoid(-(x[row,:] @ Wd[:,h] + bd[h])) = exp(-softplus(z))
// ---------------------------------------------------------------------------
__global__ __launch_bounds__(256)
void decay_kernel(const float* __restrict__ x, const float* __restrict__ Wd,
                  const float* __restrict__ bd, float* __restrict__ adec) {
  const int row   = blockIdx.x;
  const int tidx  = threadIdx.x;
  const int h     = tidx & 15;
  const int strip = tidx >> 4;       // 0..15, each 128 K values
  const float* xr = x + (size_t)row * Dd;
  float s = 0.f;
#pragma unroll 4
  for (int i = 0; i < 128; ++i) {
    int k = strip * 128 + i;
    s += xr[k] * Wd[(size_t)k * H + h];
  }
  __shared__ float red[256];
  red[tidx] = s;
  __syncthreads();
  if (strip == 0) {
    float z = bd[h];
#pragma unroll
    for (int st = 0; st < 16; ++st) z += red[st * 16 + h];
    adec[(size_t)row * H + h] = 1.f / (1.f + __expf(z));  // sigmoid(-z)
  }
}

// ---------------------------------------------------------------------------
// Recurrence pass 1: per-chunk local scan -> chunk-final state S, decay prod P
// ---------------------------------------------------------------------------
__global__ __launch_bounds__(128)
void chunk_local_kernel(const float* __restrict__ kx, const float* __restrict__ vx,
                        const float* __restrict__ gx, const float* __restrict__ adec,
                        float* __restrict__ S, float* __restrict__ P) {
  const int d = threadIdx.x, c = blockIdx.x, h = blockIdx.y, b = blockIdx.z;
  size_t base = (size_t)(b * T + c * CHUNK) * HID + h * DH + d;
  float st = 0.f, ap = 1.f;
#pragma unroll 4
  for (int t = 0; t < CHUNK; ++t) {
    size_t off = base + (size_t)t * HID;
    float a  = adec[(size_t)(b * T + c * CHUNK + t) * H + h];
    float g  = gx[off];
    float kv = (1.f / (1.f + __expf(-g))) * kx[off] * vx[off];
    st = a * st + kv;
    ap *= a;
  }
  S[((size_t)(b * NC + c) * H + h) * DH + d] = st;
  if (d == 0) P[(size_t)(b * NC + c) * H + h] = ap;
}

// ---------------------------------------------------------------------------
// Recurrence pass 2: cross-chunk scan (NC=64 steps) per (b,h,d) channel
// ---------------------------------------------------------------------------
__global__ __launch_bounds__(128)
void chunk_scan_kernel(const float* __restrict__ S, const float* __restrict__ P,
                       float* __restrict__ E) {
  const int idx = blockIdx.x * 128 + threadIdx.x;   // B*H*DH = 4096
  const int d = idx & (DH - 1);
  const int h = (idx >> 7) & (H - 1);
  const int b = idx >> 11;
  float ent = 0.f;
  for (int c = 0; c < NC; ++c) {
    size_t o = ((size_t)(b * NC + c) * H + h) * DH + d;
    E[o] = ent;
    ent  = ent * P[(size_t)(b * NC + c) * H + h] + S[o];
  }
}

// ---------------------------------------------------------------------------
// Recurrence pass 3: replay chunk with entering state, out = q * state
// (in place over q buffer).
// ---------------------------------------------------------------------------
__global__ __launch_bounds__(128)
void chunk_apply_kernel(const float* __restrict__ kx, const float* __restrict__ vx,
                        const float* __restrict__ gx, const float* __restrict__ adec,
                        const float* __restrict__ E, float* __restrict__ q_io) {
  const int d = threadIdx.x, c = blockIdx.x, h = blockIdx.y, b = blockIdx.z;
  size_t base = (size_t)(b * T + c * CHUNK) * HID + h * DH + d;
  float st = E[((size_t)(b * NC + c) * H + h) * DH + d];
#pragma unroll 4
  for (int t = 0; t < CHUNK; ++t) {
    size_t off = base + (size_t)t * HID;
    float a  = adec[(size_t)(b * T + c * CHUNK + t) * H + h];
    float g  = gx[off];
    float kv = (1.f / (1.f + __expf(-g))) * kx[off] * vx[off];
    st = a * st + kv;
    q_io[off] = q_io[off] * st;
  }
}

// ---------------------------------------------------------------------------
// Fused RMS-norm + output gate -> bf16 activation for the final GEMM.
// ---------------------------------------------------------------------------
__global__ __launch_bounds__(256)
void norm_gate_kernel(const float* __restrict__ p, const float* __restrict__ og,
                      const float* __restrict__ nw, __bf16* __restrict__ y) {
  const int row = blockIdx.x, tid = threadIdx.x;
  const float* pr = p + (size_t)row * HID;
  float ss = 0.f;
#pragma unroll
  for (int j = tid; j < HID; j += 256) { float v = pr[j]; ss += v * v; }
  __shared__ float red[256];
  red[tid] = ss;
  __syncthreads();
  for (int s = 128; s > 0; s >>= 1) {
    if (tid < s) red[tid] += red[tid + s];
    __syncthreads();
  }
  const float scale = rsqrtf(red[0] * (1.f / HID) + 1.1920929e-7f);
  const float* ogr = og + (size_t)row * HID;
  __bf16* yr = y + (size_t)row * HID;
#pragma unroll
  for (int j = tid; j < HID; j += 256) {
    float v = pr[j] * scale * nw[j] * (1.f / (1.f + __expf(-ogr[j])));
    yr[j] = (__bf16)v;
  }
}

// ---------------------------------------------------------------------------
// Host-side orchestration
// ---------------------------------------------------------------------------
extern "C" void kernel_launch(void* const* d_in, const int* in_sizes, int n_in,
                              void* d_out, int out_size, void* d_ws, size_t ws_size,
                              hipStream_t stream) {
  const float* x     = (const float*)d_in[0];
  const float* Wq    = (const float*)d_in[1];
  const float* Wk    = (const float*)d_in[2];
  const float* Wv    = (const float*)d_in[3];
  const float* Wg    = (const float*)d_in[4];
  const float* Wog   = (const float*)d_in[5];
  const float* Wo    = (const float*)d_in[6];
  const float* Wd    = (const float*)d_in[7];
  const float* bd    = (const float*)d_in[8];
  const float* normw = (const float*)d_in[9];

  // bump allocator on workspace
  char*  ws  = (char*)d_ws;
  size_t off = 0;
  auto alloc = [&](size_t bytes) -> void* {
    void* p = ws + off;
    off = (off + bytes + 255) & ~size_t(255);
    return p;
  };

  const size_t nX = (size_t)MROWS * Dd;    // 8192*2048
  const size_t nW = (size_t)Dd * HID;      // 2048*2048

  __bf16* xb   = (__bf16*)alloc(nX * sizeof(__bf16));
  __bf16* WqT  = (__bf16*)alloc(nW * sizeof(__bf16));
  __bf16* WkT  = (__bf16*)alloc(nW * sizeof(__bf16));
  __bf16* WvT  = (__bf16*)alloc(nW * sizeof(__bf16));
  __bf16* WgT  = (__bf16*)alloc(nW * sizeof(__bf16));
  __bf16* WogT = (__bf16*)alloc(nW * sizeof(__bf16));
  __bf16* WoT  = (__bf16*)alloc(nW * sizeof(__bf16));
  float*  qb   = (float*)alloc(nX * sizeof(float));
  float*  kb_  = (float*)alloc(nX * sizeof(float));
  float*  vb   = (float*)alloc(nX * sizeof(float));
  float*  gb   = (float*)alloc(nX * sizeof(float));
  float*  ogb  = (float*)alloc(nX * sizeof(float));
  float*  adec = (float*)alloc((size_t)MROWS * H * sizeof(float));
  float*  Sb   = (float*)alloc((size_t)Bb * NC * H * DH * sizeof(float));
  float*  Pb   = (float*)alloc((size_t)Bb * NC * H * sizeof(float));
  float*  Eb   = (float*)alloc((size_t)Bb * NC * H * DH * sizeof(float));
  __bf16* yb   = (__bf16*)alloc(nX * sizeof(__bf16));
  (void)ws_size; (void)n_in; (void)in_sizes; (void)out_size;

  // 1) convert x to bf16
  f32_to_bf16_kernel<<<(unsigned)((nX + 255) / 256), 256, 0, stream>>>(x, xb, nX);

  // 2) convert + transpose weights to bf16 (N x K row-major)
  dim3 tgrid(HID / 32, Dd / 32);
  transpose_f32_to_bf16_kernel<<<tgrid, 256, 0, stream>>>(Wq,  WqT,  Dd,  HID);
  transpose_f32_to_bf16_kernel<<<tgrid, 256, 0, stream>>>(Wk,  WkT,  Dd,  HID);
  transpose_f32_to_bf16_kernel<<<tgrid, 256, 0, stream>>>(Wv,  WvT,  Dd,  HID);
  transpose_f32_to_bf16_kernel<<<tgrid, 256, 0, stream>>>(Wg,  WgT,  Dd,  HID);
  transpose_f32_to_bf16_kernel<<<tgrid, 256, 0, stream>>>(Wog, WogT, Dd,  HID);
  transpose_f32_to_bf16_kernel<<<dim3(Dd / 32, HID / 32), 256, 0, stream>>>(Wo, WoT, HID, Dd);

  // 3) decay (fp32 small GEMM + softplus fused)
  decay_kernel<<<MROWS, 256, 0, stream>>>(x, Wd, bd, adec);

  // 4) big projections via bf16 WMMA GEMM (async-LDS pipelined, K=2048)
  dim3 ggrid(HID / BN, MROWS / BM);   // (16, 64)
  wmma_gemm_bf16_kernel<Dd><<<ggrid, 128, 0, stream>>>(xb, WqT,  qb,  MROWS, HID);
  wmma_gemm_bf16_kernel<Dd><<<ggrid, 128, 0, stream>>>(xb, WkT,  kb_, MROWS, HID);
  wmma_gemm_bf16_kernel<Dd><<<ggrid, 128, 0, stream>>>(xb, WvT,  vb,  MROWS, HID);
  wmma_gemm_bf16_kernel<Dd><<<ggrid, 128, 0, stream>>>(xb, WgT,  gb,  MROWS, HID);
  wmma_gemm_bf16_kernel<Dd><<<ggrid, 128, 0, stream>>>(xb, WogT, ogb, MROWS, HID);

  // 5) chunked linear recurrence (3 passes)
  chunk_local_kernel<<<dim3(NC, H, Bb), 128, 0, stream>>>(kb_, vb, gb, adec, Sb, Pb);
  chunk_scan_kernel<<<(Bb * H * DH) / 128, 128, 0, stream>>>(Sb, Pb, Eb);
  chunk_apply_kernel<<<dim3(NC, H, Bb), 128, 0, stream>>>(kb_, vb, gb, adec, Eb, qb);

  // 6) RMS-norm + output gate -> bf16
  norm_gate_kernel<<<MROWS, 256, 0, stream>>>(qb, ogb, normw, yb);

  // 7) final projection into d_out (fp32)
  wmma_gemm_bf16_kernel<HID><<<dim3(Dd / BN, MROWS / BM), 128, 0, stream>>>(
      yb, WoT, (float*)d_out, MROWS, Dd);
}
